// SpliceosomeModelWithTranscriptProbLoss_30666066494040
// MI455X (gfx1250) — matrix operand order repaired and checked by reading, not compiled, not run on hardware
//
#include <hip/hip_runtime.h>
#include <hip/hip_bf16.h>
#include <math.h>

typedef __attribute__((ext_vector_type(16))) _Float16 v16h;
typedef __attribute__((ext_vector_type(8)))  _Float16 v8h;
typedef __attribute__((ext_vector_type(8)))  float    v8f;

#define GENES   16
#define S_SITES 4096
#define CCH     512
#define JJ      4096
#define TT      64
#define KK      32
#define HH      1024
#define MROWS   32                      // junctions per block

// ---------------------------------------------------------------------------
// Prep: W [K=1024][N=1024] f32  ->  Wt [N][K] f16 (B-fragment friendly layout)
// ---------------------------------------------------------------------------
__global__ void prep_weights(const float* __restrict__ W, _Float16* __restrict__ Wt) {
    int idx = blockIdx.x * 256 + threadIdx.x;   // idx = n*1024 + k
    int n = idx >> 10;
    int k = idx & 1023;
    Wt[idx] = (_Float16)W[k * HH + n];
}

// ---------------------------------------------------------------------------
// Main: gather -> Linear(1024)+ReLU -> Linear(1024)+ReLU -> Linear(1) fused
// One block = 32 junctions of one gene. 256 threads = 8 waves (wave32).
// Phase 0: cooperative gather of x[32,1024] -> LDS (f16), once per block.
// Phase 1: layer 1 from LDS (A) x W1t (B), accumulate in regs, then overwrite
//          the SAME LDS buffer with ReLU(h1) after a barrier.
// Phase 2: layer 2 + fused layer-3 dot with W3, shfl-reduce, global atomics.
// ---------------------------------------------------------------------------
__global__ __launch_bounds__(256) void mlp_pot(
    const float* __restrict__ reps,
    const int*   __restrict__ don,
    const int*   __restrict__ acc,
    const _Float16* __restrict__ W1t, const float* __restrict__ b1,
    const _Float16* __restrict__ W2t, const float* __restrict__ b2,
    const float* __restrict__ W3,
    const float* __restrict__ gstart, const float* __restrict__ gend,
    float* __restrict__ pot)
{
    __shared__ _Float16 buf[MROWS][HH];         // 64 KB: x (phase 0/1), then h1

    const int tid  = threadIdx.x;
    const int lane = tid & 31;
    const int wv   = tid >> 5;
    const int b    = blockIdx.x >> 7;            // J/32 = 128 tiles per gene
    const int jbase = (blockIdx.x & 127) << 5;

    // ---------------- Phase 0: cooperative gather x -> LDS (f16) ----------
    {
        const int row   = tid >> 3;              // 0..31
        const int chunk = (tid & 7) << 7;        // 0,128,...,896 (channel base)
        const int j = jbase + row;
        const int idx = (chunk < CCH) ? don[b * JJ + j] : acc[b * JJ + j];
        const float* base = (idx < S_SITES)
                            ? (reps + ((size_t)b * S_SITES + idx) * CCH)
                            : ((idx == S_SITES) ? gstart : gend);
        const int coff = (chunk < CCH) ? chunk : (chunk - CCH);
#pragma unroll
        for (int c = 0; c < 128; c += 8) {
            v8f f = *(const v8f*)(base + coff + c);
            v8h h;
#pragma unroll
            for (int i = 0; i < 8; ++i) h[i] = (_Float16)f[i];
            *(v8h*)(&buf[row][chunk + c]) = h;   // ds_store_b128
        }
    }
    __syncthreads();

    const int m    = lane & 15;
    const int off  = (lane < 16) ? 0 : 8;     // A-fragment K half select (2*off, contiguous, matches B)
    const int kbB  = (lane < 16) ? 0 : 16;    // B-fragment K half select
    const int n0w  = wv * 128;                // this wave's column range
    const int mrow = (lane < 16) ? 0 : 8;     // C/D fragment row base

    const _Float16* hrow0 = &buf[m][0];
    const _Float16* hrow1 = &buf[16 + m][0];

    // ---------------- Phase 1: layer 1  x[32,1024] @ W1 ----------------
    v8f acc1[2][8] = {};
    for (int k0 = 0; k0 < HH; k0 += 32) {
        union { v16h v; v8h h[2]; } A[2];
        A[0].h[0] = *(const v8h*)(hrow0 + k0 + 2 * off);        // ds_load_b128
        A[0].h[1] = *(const v8h*)(hrow0 + k0 + 8 + 2 * off);
        A[1].h[0] = *(const v8h*)(hrow1 + k0 + 2 * off);
        A[1].h[1] = *(const v8h*)(hrow1 + k0 + 8 + 2 * off);

        __builtin_prefetch(W1t + (size_t)(n0w + m) * HH + k0 + kbB + 32, 0, 1);
#pragma unroll
        for (int nt = 0; nt < 8; ++nt) {
            const _Float16* bp = W1t + (size_t)(n0w + nt * 16 + m) * HH + (k0 + kbB);
            v16h B = *(const v16h*)bp;                // one B load feeds 2 WMMAs
            acc1[0][nt] = __builtin_amdgcn_wmma_f32_16x16x32_f16(
                false, A[0].v, false, B, (short)0, acc1[0][nt], false, false);
            acc1[1][nt] = __builtin_amdgcn_wmma_f32_16x16x32_f16(
                false, A[1].v, false, B, (short)0, acc1[1][nt], false, false);
        }
    }
    __syncthreads();      // all waves done READING x before we overwrite buf

    // epilogue: bias + ReLU -> h1 overwrites buf (f16)
#pragma unroll
    for (int nt = 0; nt < 8; ++nt) {
        const int n = n0w + nt * 16 + m;
        const float bias = b1[n];
#pragma unroll
        for (int mt = 0; mt < 2; ++mt) {
#pragma unroll
            for (int r = 0; r < 8; ++r) {
                float v = acc1[mt][nt][r] + bias;
                v = v > 0.0f ? v : 0.0f;
                buf[mt * 16 + mrow + r][n] = (_Float16)v;
            }
        }
    }
    __syncthreads();

    // ---------------- Phase 2: layer 2 (+ fused layer 3) ----------------
    v8f acc2[2][8] = {};
    for (int k0 = 0; k0 < HH; k0 += 32) {
        union { v16h v; v8h h[2]; } A[2];
        A[0].h[0] = *(const v8h*)(hrow0 + k0 + 2 * off);
        A[0].h[1] = *(const v8h*)(hrow0 + k0 + 8 + 2 * off);
        A[1].h[0] = *(const v8h*)(hrow1 + k0 + 2 * off);
        A[1].h[1] = *(const v8h*)(hrow1 + k0 + 8 + 2 * off);

        __builtin_prefetch(W2t + (size_t)(n0w + m) * HH + k0 + kbB + 32, 0, 1);
#pragma unroll
        for (int nt = 0; nt < 8; ++nt) {
            const _Float16* bp = W2t + (size_t)(n0w + nt * 16 + m) * HH + (k0 + kbB);
            v16h B = *(const v16h*)bp;
            acc2[0][nt] = __builtin_amdgcn_wmma_f32_16x16x32_f16(
                false, A[0].v, false, B, (short)0, acc2[0][nt], false, false);
            acc2[1][nt] = __builtin_amdgcn_wmma_f32_16x16x32_f16(
                false, A[1].v, false, B, (short)0, acc2[1][nt], false, false);
        }
    }
    // epilogue: bias + ReLU, multiply by W3 column, per-lane partial row sums
    float psum[2][8];
#pragma unroll
    for (int mt = 0; mt < 2; ++mt)
#pragma unroll
        for (int r = 0; r < 8; ++r) psum[mt][r] = 0.0f;
#pragma unroll
    for (int nt = 0; nt < 8; ++nt) {
        const int n = n0w + nt * 16 + m;
        const float bias = b2[n];
        const float w3n  = W3[n];
#pragma unroll
        for (int mt = 0; mt < 2; ++mt) {
#pragma unroll
            for (int r = 0; r < 8; ++r) {
                float v = acc2[mt][nt][r] + bias;
                v = v > 0.0f ? v : 0.0f;
                psum[mt][r] += v * w3n;
            }
        }
    }
    // cross-lane reduce within each 16-lane half (columns), then global atomic
#pragma unroll
    for (int mt = 0; mt < 2; ++mt) {
#pragma unroll
        for (int r = 0; r < 8; ++r) {
            float v = psum[mt][r];
            v += __shfl_xor(v, 1, 32);
            v += __shfl_xor(v, 2, 32);
            v += __shfl_xor(v, 4, 32);
            v += __shfl_xor(v, 8, 32);
            if ((lane & 15) == 0)
                atomicAdd(&pot[b * JJ + jbase + mt * 16 + mrow + r], v);
        }
    }
}

// ---------------------------------------------------------------------------
// Finalize: per-gene transcript sums + softmax over T+1 = 65 potentials
// ---------------------------------------------------------------------------
__global__ void finalize(const float* __restrict__ pot, const int* __restrict__ tj,
                         const float* __restrict__ refp, const float* __restrict__ b3,
                         float* __restrict__ out)
{
    const int b = blockIdx.x;
    const int t = threadIdx.x;
    __shared__ float ap[TT + 1];
    __shared__ float red[2];

    if (t < TT) {
        float s = 0.0f;
        const int* p = tj + ((size_t)b * TT + t) * KK;
        for (int k = 0; k < KK; ++k) s += pot[b * JJ + p[k]];
        ap[t] = s + (float)KK * b3[0];           // per-junction bias folded in
    }
    if (t == TT) ap[TT] = refp[0];
    __syncthreads();
    if (t == 0) {
        float mx = ap[0];
        for (int i = 1; i <= TT; ++i) mx = fmaxf(mx, ap[i]);
        float sum = 0.0f;
        for (int i = 0; i <= TT; ++i) sum += __expf(ap[i] - mx);
        red[0] = mx; red[1] = sum;
    }
    __syncthreads();
    if (t <= TT) out[b * (TT + 1) + t] = __expf(ap[t] - red[0]) / red[1];
}

// ---------------------------------------------------------------------------
extern "C" void kernel_launch(void* const* d_in, const int* in_sizes, int n_in,
                              void* d_out, int out_size, void* d_ws, size_t ws_size,
                              hipStream_t stream)
{
    const float* reps   = (const float*)d_in[0];
    const int*   don    = (const int*)d_in[1];
    const int*   acc    = (const int*)d_in[2];
    const int*   tj     = (const int*)d_in[3];
    const float* W1     = (const float*)d_in[4];
    const float* b1     = (const float*)d_in[5];
    const float* W2     = (const float*)d_in[6];
    const float* b2     = (const float*)d_in[7];
    const float* W3     = (const float*)d_in[8];
    const float* b3     = (const float*)d_in[9];
    const float* gstart = (const float*)d_in[10];
    const float* gend   = (const float*)d_in[11];
    const float* refp   = (const float*)d_in[12];

    char* ws = (char*)d_ws;
    _Float16* W1t = (_Float16*)ws;                              // 2 MB
    _Float16* W2t = (_Float16*)(ws + (size_t)HH * HH * 2);      // 2 MB
    float*    pot = (float*)   (ws + (size_t)HH * HH * 4);      // 256 KB

    prep_weights<<<HH * HH / 256, 256, 0, stream>>>(W1, W1t);
    prep_weights<<<HH * HH / 256, 256, 0, stream>>>(W2, W2t);

    hipMemsetAsync(pot, 0, (size_t)GENES * JJ * sizeof(float), stream);

    mlp_pot<<<GENES * JJ / MROWS, 256, 0, stream>>>(
        reps, don, acc, W1t, b1, W2t, b2, W3, gstart, gend, pot);

    finalize<<<GENES, 96, 0, stream>>>(pot, tj, refp, b3, (float*)d_out);
}
